// BiLSTMCRF_26654567039078
// MI455X (gfx1250) — matrix operand order repaired and checked by reading, not compiled, not run on hardware
//
#include <hip/hip_runtime.h>
#include <hip/hip_bf16.h>
#include <math.h>

#define NEGV     (-1000.0f)
#define BB       32
#define SS       512
#define EE       768
#define HHD      384
#define G4       1536      // 4*HHD
#define NTAG     17
#define STARTTAG 15
#define STOPTAG  16
#define NTILE    96        // G4/16
#define KB_IH    24        // EE/32
#define KB_HH    12        // HHD/32
#define MT_IN    1024      // (BB*SS)/16
#define RBLK     24        // recurrent blocks per direction (192 waves = 2x96 tiles)

typedef _Float16 f16x8  __attribute__((ext_vector_type(8)));
typedef _Float16 f16x16 __attribute__((ext_vector_type(16)));
typedef float    f32x8  __attribute__((ext_vector_type(8)));

__device__ __forceinline__ float sigm(float x) { return 1.0f / (1.0f + expf(-x)); }

__device__ __forceinline__ f16x16 cat8(f16x8 lo, f16x8 hi) {
    return __builtin_shufflevector(lo, hi, 0,1,2,3,4,5,6,7,8,9,10,11,12,13,14,15);
}

// ---------------- weight pre-pack: W[G4 x K] fp32 -> B-fragments f16 -------
// layout: [dir][ntile][kbi][lane(n=lane%16)][h(k table order)]
__global__ void k_pack_w(const float* __restrict__ Wf, const float* __restrict__ Wb,
                         _Float16* __restrict__ out, int K, int KB) {
    int idx = blockIdx.x * blockDim.x + threadIdx.x;
    int per_dir = NTILE * KB * 512;
    if (idx >= 2 * per_dir) return;
    int dir = idx / per_dir;
    int r   = idx % per_dir;
    int h    = r & 15;
    int lane = (r >> 4) & 31;
    int q    = r >> 9;
    int kbi  = q % KB;
    int nt   = q / KB;
    int n = nt * 16 + (lane & 15);
    int k = kbi * 32 + (h < 8 ? ((lane >> 4) << 3) + h : 16 + ((lane >> 4) << 3) + (h - 8));
    const float* W = dir ? Wb : Wf;
    out[idx] = (_Float16)W[(size_t)n * K + k];
}

__global__ void k_cvt_x(const float* __restrict__ x, _Float16* __restrict__ o, int n) {
    int i = blockIdx.x * blockDim.x + threadIdx.x;
    if (i < n) o[i] = (_Float16)x[i];
}

__global__ void k_init_state(const float* __restrict__ h0, const float* __restrict__ c0,
                             _Float16* __restrict__ hcur, float* __restrict__ cbuf) {
    int i = blockIdx.x * blockDim.x + threadIdx.x;
    if (i < 2 * BB * HHD) { hcur[i] = (_Float16)h0[i]; cbuf[i] = c0[i]; }
}

// ------------- big input GEMM: Gpre[dir][b*S+s][1536] = X @ Wih^T + bias ----
__global__ __launch_bounds__(256)
void k_gemm_in(const _Float16* __restrict__ Xf16, const _Float16* __restrict__ WihPack,
               const float* __restrict__ bias_f, const float* __restrict__ bias_b,
               _Float16* __restrict__ Gpre) {
    int wave = blockIdx.x * (blockDim.x >> 5) + (threadIdx.x >> 5);
    int lane = threadIdx.x & 31;
    int nt  = wave % NTILE;
    int mt  = (wave / NTILE) % MT_IN;
    int dir = wave / (NTILE * MT_IN);

    const _Float16* pack = WihPack + ((size_t)(dir * NTILE + nt)) * KB_IH * 512;
    int row  = mt * 16 + (lane & 15);
    int koff = (lane >> 4) << 3;
    const _Float16* arow = Xf16 + (size_t)row * EE + koff;

    f32x8 acc = {};
    for (int kbi = 0; kbi < KB_IH; ++kbi) {
        f16x8 lo = *(const f16x8*)(arow + kbi * 32);
        f16x8 hi = *(const f16x8*)(arow + kbi * 32 + 16);
        f16x16 a = cat8(lo, hi);
        f16x16 b = *(const f16x16*)(pack + (size_t)kbi * 512 + lane * 16);
        acc = __builtin_amdgcn_wmma_f32_16x16x32_f16(false, a, false, b, (short)0, acc, false, false);
    }
    int n = nt * 16 + (lane & 15);
    float bias = (dir ? bias_b : bias_f)[n];
    _Float16* gout = Gpre + (size_t)dir * BB * SS * G4;
    int mbase = mt * 16 + ((lane >> 4) << 3);
#pragma unroll
    for (int r = 0; r < 8; ++r) {
        int rr = mbase + r;                       // rr = b*SS + s
        gout[(size_t)rr * G4 + n] = (_Float16)(acc[r] + bias);
    }
}

// ---------------- grid barrier (persistent recurrent kernel) ---------------
__device__ __forceinline__ void gbar(unsigned* cnt, unsigned expected) {
    __threadfence();
    __syncthreads();
    if (threadIdx.x == 0) {
        __hip_atomic_fetch_add(cnt, 1u, __ATOMIC_ACQ_REL, __HIP_MEMORY_SCOPE_AGENT);
        while (__hip_atomic_load(cnt, __ATOMIC_ACQUIRE, __HIP_MEMORY_SCOPE_AGENT) < expected)
            __builtin_amdgcn_s_sleep(1);
    }
    __syncthreads();
    __threadfence();
}

// ------------- recurrent LSTM: 512 steps, both directions in one grid ------
__global__ __launch_bounds__(256)
void k_recur(const _Float16* __restrict__ Gpre, const _Float16* __restrict__ WhhPack,
             _Float16* __restrict__ hcur_all, float* __restrict__ cbuf_all,
             float* __restrict__ gscr_all, float* __restrict__ hseq_all,
             unsigned* __restrict__ bar) {
    const int dir = blockIdx.x / RBLK;
    const int blk = blockIdx.x % RBLK;
    const int tid = threadIdx.x;
    const int lane = tid & 31;
    const int wv = blk * 8 + (tid >> 5);       // 0..191
    const int mt = wv / NTILE;                 // 0..1
    const int nt = wv % NTILE;                 // 0..95

    const _Float16* Gd = Gpre + (size_t)dir * BB * SS * G4;
    const _Float16* Wd = WhhPack + (size_t)dir * NTILE * KB_HH * 512;
    _Float16* hcur = hcur_all + (size_t)dir * BB * HHD;
    float* cb   = cbuf_all + (size_t)dir * BB * HHD;
    float* gs   = gscr_all + (size_t)dir * BB * G4;
    float* hseq = hseq_all + (size_t)dir * SS * BB * HHD;
    unsigned* cnt = bar + dir;
    unsigned expected = 0;

    const int arow_r = mt * 16 + (lane & 15);
    const int koff   = (lane >> 4) << 3;
    const int ncol   = nt * 16 + (lane & 15);
    const int mbase  = mt * 16 + ((lane >> 4) << 3);
    const int egid   = blk * 256 + tid;        // 0..6143

    for (int step = 0; step < SS; ++step) {
        const int s = dir ? (SS - 1 - step) : step;

        // stage 1: g = Gpre[:,s,:] + h @ Whh^T  (WMMA, f32 accum)
        f32x8 acc = {};
        const _Float16* arow = hcur + (size_t)arow_r * HHD + koff;
        for (int kbi = 0; kbi < KB_HH; ++kbi) {
            f16x8 lo = *(const f16x8*)(arow + kbi * 32);
            f16x8 hi = *(const f16x8*)(arow + kbi * 32 + 16);
            f16x16 a = cat8(lo, hi);
            f16x16 b = *(const f16x16*)(Wd + ((size_t)nt * KB_HH + kbi) * 512 + lane * 16);
            acc = __builtin_amdgcn_wmma_f32_16x16x32_f16(false, a, false, b, (short)0, acc, false, false);
        }
#pragma unroll
        for (int r = 0; r < 8; ++r) {
            int bq = mbase + r;
            float g = acc[r] + (float)Gd[((size_t)bq * SS + s) * G4 + ncol];
            gs[(size_t)bq * G4 + ncol] = g;
        }
        expected += RBLK;
        gbar(cnt, expected);

        // stage 2: elementwise LSTM cell
        for (int e = egid; e < BB * HHD; e += RBLK * 256) {
            int b = e / HHD, j = e % HHD;
            float gi = gs[(size_t)b * G4 + j];
            float gf = gs[(size_t)b * G4 + HHD + j];
            float gg = gs[(size_t)b * G4 + 2 * HHD + j];
            float go = gs[(size_t)b * G4 + 3 * HHD + j];
            float c = sigm(gf) * cb[e] + sigm(gi) * tanhf(gg);
            float h = sigm(go) * tanhf(c);
            cb[e] = c;
            hcur[e] = (_Float16)h;
            hseq[((size_t)s * BB + b) * HHD + j] = h;
        }
        expected += RBLK;
        gbar(cnt, expected);
    }
}

// ---------------- tag projection: logits[b][s][t] ---------------------------
__global__ void k_logits(const float* __restrict__ hseq_all, const float* __restrict__ Wtag,
                         const float* __restrict__ btag, float* __restrict__ logits) {
    int gid = blockIdx.x * blockDim.x + threadIdx.x;
    if (gid >= BB * SS * NTAG) return;
    int t = gid % NTAG;
    int s = (gid / NTAG) % SS;
    int b = gid / (NTAG * SS);
    const float* hf = hseq_all + ((size_t)s * BB + b) * HHD;
    const float* hb = hseq_all + (size_t)SS * BB * HHD + ((size_t)s * BB + b) * HHD;
    const float* w = Wtag + (size_t)t * EE;
    float acc = btag[t];
#pragma unroll 4
    for (int j = 0; j < HHD; ++j) acc += hf[j] * w[j] + hb[j] * w[HHD + j];
    logits[((size_t)b * SS + s) * NTAG + t] = acc;
}

// ---------------- Viterbi / CRF decode, one wave per batch ------------------
__global__ __launch_bounds__(32)
void k_viterbi(const float* __restrict__ logits, const int* __restrict__ mask,
               const float* __restrict__ trans, int* __restrict__ bps,
               float* __restrict__ out) {
    int b = blockIdx.x;
    int lane = threadIdx.x;
    __shared__ float tr[NTAG * NTAG];
    __shared__ float alpha[NTAG];
    __shared__ int len_s;
    for (int i = lane; i < NTAG * NTAG; i += 32) tr[i] = trans[i];
    if (lane == 0) {
        int L = 0;
        for (int s = 0; s < SS; ++s) L += mask[b * SS + s];
        len_s = L;
    }
    if (lane < NTAG) alpha[lane] = (lane == STARTTAG) ? 0.0f : NEGV;
    __syncthreads();
    const int len = len_s;
    const float* lg = logits + (size_t)b * SS * NTAG;
    int* bpb = bps + (size_t)b * SS * NTAG;

    for (int s = 0; s < SS; ++s) {
        float na = 0.0f;
        if (lane < NTAG) {
            float fr = lg[s * NTAG + lane];
            float sv[NTAG];
            float m = -3.0e38f; int bp = 0;
#pragma unroll
            for (int p = 0; p < NTAG; ++p) {
                float v = alpha[p] + fr + tr[p * NTAG + lane];
                sv[p] = v;
                if (v > m) { m = v; bp = p; }
            }
            float sum = 0.0f;
#pragma unroll
            for (int p = 0; p < NTAG; ++p) sum += expf(sv[p] - m);
            na = m + logf(sum);
            bpb[s * NTAG + lane] = bp;
            if (s >= len) na = alpha[lane];
        }
        __syncthreads();
        if (lane < NTAG) alpha[lane] = na;
        __syncthreads();
    }
    if (lane == 0) {
        float fin[NTAG];
        float m = -3.0e38f; int last = 0;
#pragma unroll
        for (int t = 0; t < NTAG; ++t) {
            fin[t] = alpha[t] + tr[t * NTAG + STOPTAG];
            if (fin[t] > m) { m = fin[t]; last = t; }
        }
        float sum = 0.0f;
#pragma unroll
        for (int t = 0; t < NTAG; ++t) sum += expf(fin[t] - m);
        out[b] = m + logf(sum);
        int tag = last;
        float* path = out + BB + (size_t)b * SS;
        for (int t = SS - 1; t >= 0; --t) {
            int nt;
            if (t == len - 1)      nt = last;
            else if (t < len - 1)  nt = bpb[(t + 1) * NTAG + tag];
            else                   nt = tag;
            tag = nt;
            path[t] = (t < len) ? (float)nt : -1.0f;
        }
    }
}

extern "C" void kernel_launch(void* const* d_in, const int* in_sizes, int n_in,
                              void* d_out, int out_size, void* d_ws, size_t ws_size,
                              hipStream_t stream) {
    (void)in_sizes; (void)n_in; (void)out_size; (void)ws_size;
    const float* emb   = (const float*)d_in[0];
    const int*   mask  = (const int*)d_in[1];
    const float* h0    = (const float*)d_in[2];
    const float* c0    = (const float*)d_in[3];
    const float* Wih_f = (const float*)d_in[4];
    const float* Whh_f = (const float*)d_in[5];
    const float* b_f   = (const float*)d_in[6];
    const float* Wih_b = (const float*)d_in[7];
    const float* Whh_b = (const float*)d_in[8];
    const float* b_b   = (const float*)d_in[9];
    const float* Wtag  = (const float*)d_in[10];
    const float* btag  = (const float*)d_in[11];
    const float* trans = (const float*)d_in[12];

    char* ws = (char*)d_ws;
    size_t off = 0;
    auto take = [&](size_t bytes) {
        char* p = ws + off;
        off += (bytes + 255) & ~(size_t)255;
        return p;
    };
    unsigned*  bar     = (unsigned*)take(8);
    _Float16*  WihPack = (_Float16*)take((size_t)2 * NTILE * KB_IH * 512 * 2);
    _Float16*  WhhPack = (_Float16*)take((size_t)2 * NTILE * KB_HH * 512 * 2);
    _Float16*  Xf16    = (_Float16*)take((size_t)BB * SS * EE * 2);
    _Float16*  Gpre    = (_Float16*)take((size_t)2 * BB * SS * G4 * 2);
    float*     hseq    = (float*)take((size_t)2 * SS * BB * HHD * 4);
    _Float16*  hcur    = (_Float16*)take((size_t)2 * BB * HHD * 2);
    float*     cbuf    = (float*)take((size_t)2 * BB * HHD * 4);
    float*     gscr    = (float*)take((size_t)2 * BB * G4 * 4);
    float*     logits  = (float*)take((size_t)BB * SS * NTAG * 4);
    int*       bps     = (int*)take((size_t)BB * SS * NTAG * 4);

    hipMemsetAsync(bar, 0, 8, stream);

    int n1 = 2 * NTILE * KB_IH * 512;
    k_pack_w<<<(n1 + 255) / 256, 256, 0, stream>>>(Wih_f, Wih_b, WihPack, EE, KB_IH);
    int n2 = 2 * NTILE * KB_HH * 512;
    k_pack_w<<<(n2 + 255) / 256, 256, 0, stream>>>(Whh_f, Whh_b, WhhPack, HHD, KB_HH);
    int n3 = BB * SS * EE;
    k_cvt_x<<<(n3 + 255) / 256, 256, 0, stream>>>(emb, Xf16, n3);
    k_init_state<<<(2 * BB * HHD + 255) / 256, 256, 0, stream>>>(h0, c0, hcur, cbuf);

    int nwaves = 2 * MT_IN * NTILE;               // 196608 tiles, 1 wave each
    k_gemm_in<<<nwaves / 8, 256, 0, stream>>>(Xf16, WihPack, b_f, b_b, Gpre);

    k_recur<<<2 * RBLK, 256, 0, stream>>>(Gpre, WhhPack, hcur, cbuf, gscr, hseq, bar);

    int n4 = BB * SS * NTAG;
    k_logits<<<(n4 + 255) / 256, 256, 0, stream>>>(hseq, Wtag, btag, logits);
    k_viterbi<<<BB, 32, 0, stream>>>(logits, mask, trans, bps, (float*)d_out);
}
